// HyperAttention_27247272526197
// MI455X (gfx1250) — compile-verified, matrix operand order checked by reading, not run on hardware
//
#include <hip/hip_runtime.h>

#define Bb 2
#define Hh 16
#define NN 8192
#define DD 64
#define RPROJ 7
#define BLK 256
#define SMP 256
#define BH_ (Bb*Hh)
#define BHN_ (BH_*NN)
#define SCALE_ 0.125f
#define LOG32_ 3.4657359027997265f

typedef __attribute__((ext_vector_type(16))) _Float16     v16h;
typedef __attribute__((ext_vector_type(8)))  float        v8f;
typedef __attribute__((ext_vector_type(4)))  _Float16     h4;
typedef __attribute__((ext_vector_type(4)))  float        f4;
typedef __attribute__((ext_vector_type(4)))  unsigned int u32x4;

union F16Frag { v16h v; u32x4 u[2]; };

// Load one WMMA 16-bit fragment: per CDNA5 layout, lane (kh = lane>>4) needs
// halfs [base .. base+7] (j=0..7) and [base+16 .. base+23] (j=8..15), i.e. two
// contiguous 16-byte loads.
__device__ __forceinline__ v16h load_frag(const _Float16* p) {
  F16Frag f;
  f.u[0] = *(const u32x4*)(p);
  f.u[1] = *(const u32x4*)(p + 16);
  return f.v;
}

#define WMMA_F16(A, B, C) \
  __builtin_amdgcn_wmma_f32_16x16x32_f16(false, (A), false, (B), (short)0, (C), false, false)

// ---------------------------------------------------------------------------
// Kernel 1: LSH hash (gray-coded bucket) for queries and keys.
// grid: 2*BHN_/256 blocks x 256 threads
// ---------------------------------------------------------------------------
__global__ void hash_kernel(const float* __restrict__ q, const float* __restrict__ k,
                            const float* __restrict__ pd,
                            int* __restrict__ qh, int* __restrict__ kh) {
  __shared__ float pds[DD * RPROJ];
  for (int i = threadIdx.x; i < DD * RPROJ; i += blockDim.x) pds[i] = pd[i];
  __syncthreads();

  long long t = (long long)blockIdx.x * blockDim.x + threadIdx.x;
  int isK = 0;
  long long r = t;
  if (r >= (long long)BHN_) { isK = 1; r -= BHN_; }
  const float* src = (isK ? k : q) + r * DD;

  float acc[RPROJ];
#pragma unroll
  for (int j = 0; j < RPROJ; j++) acc[j] = 0.0f;
#pragma unroll
  for (int d = 0; d < DD; d++) {
    float x = src[d];
#pragma unroll
    for (int j = 0; j < RPROJ; j++) acc[j] += x * pds[d * RPROJ + j];
  }
  int bin = 0;
#pragma unroll
  for (int j = 0; j < RPROJ; j++) bin |= (acc[j] > 0.0f) ? (1 << j) : 0;
  int bucket = bin ^ (bin >> 1);  // gray-code permutation
  (isK ? kh : qh)[r] = bucket;
}

// ---------------------------------------------------------------------------
// Kernel 2: stable counting sort (keys in [0,128)) per (b,h), for q and k.
// grid: 2*BH_ blocks x 128 threads. Each thread owns 64 contiguous elements.
// ---------------------------------------------------------------------------
__global__ void csort_kernel(const int* __restrict__ qh, const int* __restrict__ kh,
                             int* __restrict__ qi, int* __restrict__ ki) {
  __shared__ unsigned short cnt[128 * 128];  // [key][thread], 32 KB
  __shared__ unsigned int   part[128];
  int which = blockIdx.x >> 5;  // BH_ == 32
  int bh    = blockIdx.x & 31;
  const int* h   = (which ? kh : qh) + (long long)bh * NN;
  int*       out = (which ? ki : qi) + (long long)bh * NN;
  int t = threadIdx.x;

#pragma unroll 8
  for (int i = 0; i < 128; i++) cnt[t * 128 + i] = 0;
  __syncthreads();

  const int EPT = NN / 128;  // 64
  for (int e = 0; e < EPT; e++) {
    int key = h[t * EPT + e];
    cnt[key * 128 + t] = (unsigned short)(cnt[key * 128 + t] + 1);
  }
  __syncthreads();

  // exclusive scan over flattened [key*128 + thread] order (key-major => stable)
  unsigned int run = 0;
  for (int j = 0; j < 128; j++) {
    unsigned int v = cnt[t * 128 + j];
    cnt[t * 128 + j] = (unsigned short)run;
    run += v;
  }
  part[t] = run;
  __syncthreads();
  if (t == 0) {
    unsigned int r2 = 0;
    for (int j = 0; j < 128; j++) { unsigned int v = part[j]; part[j] = r2; r2 += v; }
  }
  __syncthreads();
  unsigned int base = part[t];
  for (int j = 0; j < 128; j++)
    cnt[t * 128 + j] = (unsigned short)(cnt[t * 128 + j] + base);
  __syncthreads();

  for (int e = 0; e < EPT; e++) {
    int i = t * EPT + e;
    int key = h[i];
    int pos = cnt[key * 128 + t];
    cnt[key * 128 + t] = (unsigned short)(pos + 1);
    out[(long long)pos] = i;
  }
}

// ---------------------------------------------------------------------------
// Kernel 3: gather sorted q/k/v rows, convert f32 -> f16.
// grid: 3*BHN_/256 blocks x 256 threads; one thread per 64-element row.
// ---------------------------------------------------------------------------
__global__ void gather_kernel(const float* __restrict__ q, const float* __restrict__ k,
                              const float* __restrict__ v,
                              const int* __restrict__ qi, const int* __restrict__ ki,
                              _Float16* __restrict__ qs, _Float16* __restrict__ ks,
                              _Float16* __restrict__ vs) {
  long long t = (long long)blockIdx.x * blockDim.x + threadIdx.x;
  int sel = (int)(t / BHN_);
  long long r = t % BHN_;
  long long bh = r / NN;
  int p = ((sel == 0) ? qi : ki)[r];
  const float* src = ((sel == 0) ? q : (sel == 1) ? k : v) + (bh * NN + p) * DD;
  _Float16* dst = ((sel == 0) ? qs : (sel == 1) ? ks : vs) + r * DD;
  const f4* s4 = (const f4*)src;
  h4* d4 = (h4*)dst;
#pragma unroll
  for (int c = 0; c < 16; c++) {
    f4 f = s4[c];
    h4 o;
    o[0] = (_Float16)f[0]; o[1] = (_Float16)f[1];
    o[2] = (_Float16)f[2]; o[3] = (_Float16)f[3];
    d4[c] = o;
  }
}

// ---------------------------------------------------------------------------
// Kernel 4: block-diagonal attention over sorted order.
// grid: BH_*32*2 blocks x 256 threads (8 waves; each wave = 16 query rows).
// Writes attn_block scattered to d_out (un-sorted rows) + lse (sorted order).
// ---------------------------------------------------------------------------
__global__ __launch_bounds__(256) void blockattn_kernel(
    const _Float16* __restrict__ qs, const _Float16* __restrict__ ks,
    const _Float16* __restrict__ vs, const int* __restrict__ qi,
    float* __restrict__ lseW, float* __restrict__ out) {
  __shared__ _Float16 Kt[BLK * DD];  // [krow][d]   32 KB (reused for P staging)
  __shared__ _Float16 Vt[DD * BLK];  // [d][krow]   32 KB (transposed)

  int wg = blockIdx.x;
  int bh = wg >> 6;
  int rem = wg & 63;
  int blk = rem >> 1;
  int half = rem & 1;
  long long bhN = (long long)bh * NN;
  int t = threadIdx.x;

  {  // cooperative tile load: thread t handles key-row blk*256+t
    long long krow = bhN + blk * BLK + t;
    const u32x4* s4 = (const u32x4*)(ks + krow * DD);
#pragma unroll
    for (int i = 0; i < 8; i++) *(u32x4*)(&Kt[t * DD + i * 8]) = s4[i];
    const unsigned int* s32 = (const unsigned int*)(vs + krow * DD);
    unsigned short* VtU = (unsigned short*)Vt;
#pragma unroll
    for (int i = 0; i < 32; i++) {
      unsigned int w = s32[i];
      VtU[(2 * i + 0) * BLK + t] = (unsigned short)(w & 0xffffu);
      VtU[(2 * i + 1) * BLK + t] = (unsigned short)(w >> 16);
    }
  }
  __syncthreads();

  int wave = t >> 5, lane = t & 31;
  int ln = lane & 15, kh2 = lane >> 4;
  int rowbase = blk * BLK + half * 128 + wave * 16;

  const _Float16* qrow = qs + (bhN + rowbase + ln) * DD;
  v16h qa0 = load_frag(qrow + kh2 * 8);
  v16h qa1 = load_frag(qrow + 32 + kh2 * 8);

  v8f S[16];
#pragma unroll
  for (int n = 0; n < 16; n++) {
    v16h b0 = load_frag(&Kt[(n * 16 + ln) * DD + kh2 * 8]);
    v16h b1 = load_frag(&Kt[(n * 16 + ln) * DD + 32 + kh2 * 8]);
    v8f acc = {};
    acc = WMMA_F16(qa0, b0, acc);
    acc = WMMA_F16(qa1, b1, acc);
    S[n] = acc;
  }

  float mx[8], sm[8];
#pragma unroll
  for (int r = 0; r < 8; r++) {
    float m = -3.0e38f;
#pragma unroll
    for (int n = 0; n < 16; n++) m = fmaxf(m, S[n][r]);
#pragma unroll
    for (int o = 1; o < 16; o <<= 1) m = fmaxf(m, __shfl_xor(m, o, 32));
    mx[r] = m;
  }
#pragma unroll
  for (int r = 0; r < 8; r++) {
    float s = 0.0f;
#pragma unroll
    for (int n = 0; n < 16; n++) {
      float p = __expf((S[n][r] - mx[r]) * SCALE_);
      S[n][r] = p;
      s += p;
    }
#pragma unroll
    for (int o = 1; o < 16; o <<= 1) s += __shfl_xor(s, o, 32);
    sm[r] = s;
  }
  if (ln == 0) {
#pragma unroll
    for (int r = 0; r < 8; r++)
      lseW[bhN + rowbase + kh2 * 8 + r] = mx[r] * SCALE_ + __logf(sm[r]);
  }

  __syncthreads();  // all waves done reading Kt -> reuse as per-wave P staging
  _Float16* Pst = Kt + wave * 512;  // 16x32 halfs per wave

  v8f O[4] = {};
#pragma unroll
  for (int kc = 0; kc < 8; kc++) {
#pragma unroll
    for (int u = 0; u < 2; u++) {
      int n = kc * 2 + u;
#pragma unroll
      for (int r = 0; r < 8; r++)
        Pst[(r + kh2 * 8) * 32 + u * 16 + ln] = (_Float16)S[n][r];
    }
    __builtin_amdgcn_wave_barrier();  // LDS is in-order per wave; block re-sched
    v16h pa = load_frag(&Pst[ln * 32 + kh2 * 8]);
#pragma unroll
    for (int nv = 0; nv < 4; nv++) {
      v16h vb = load_frag(&Vt[(nv * 16 + ln) * BLK + kc * 32 + kh2 * 8]);
      O[nv] = WMMA_F16(pa, vb, O[nv]);
    }
    __builtin_amdgcn_wave_barrier();
  }

  float rs[8];
  int ur[8];
#pragma unroll
  for (int r = 0; r < 8; r++) {
    rs[r] = 1.0f / sm[r];
    ur[r] = qi[bhN + rowbase + kh2 * 8 + r];
  }
#pragma unroll
  for (int nv = 0; nv < 4; nv++)
#pragma unroll
    for (int r = 0; r < 8; r++)
      out[(bhN + ur[r]) * DD + nv * 16 + ln] = O[nv][r] * rs[r];
}

// ---------------------------------------------------------------------------
// Kernel 5: sampled residual attention + online-softmax merge + final write.
// grid: BH_*64 blocks x 256 threads (128 query rows per block).
// ---------------------------------------------------------------------------
__global__ __launch_bounds__(256) void residual_kernel(
    const _Float16* __restrict__ qs, const _Float16* __restrict__ ks,
    const _Float16* __restrict__ vs, const int* __restrict__ qi,
    const int* __restrict__ smp, const float* __restrict__ lseW,
    float* __restrict__ out) {
  __shared__ _Float16 Kt[SMP * DD];
  __shared__ _Float16 Vt[DD * SMP];

  int wg = blockIdx.x;
  int bh = wg >> 6;
  int tile = wg & 63;
  int blk = tile >> 1;  // 256-block id of all rows in this workgroup
  long long bhN = (long long)bh * NN;
  int t = threadIdx.x;

  {  // gather sampled key/value rows (already sorted-space indices)
    int sidx = smp[bh * SMP + t];
    long long krow = bhN + sidx;
    const u32x4* s4 = (const u32x4*)(ks + krow * DD);
#pragma unroll
    for (int i = 0; i < 8; i++) *(u32x4*)(&Kt[t * DD + i * 8]) = s4[i];
    const unsigned int* s32 = (const unsigned int*)(vs + krow * DD);
    unsigned short* VtU = (unsigned short*)Vt;
#pragma unroll
    for (int i = 0; i < 32; i++) {
      unsigned int w = s32[i];
      VtU[(2 * i + 0) * SMP + t] = (unsigned short)(w & 0xffffu);
      VtU[(2 * i + 1) * SMP + t] = (unsigned short)(w >> 16);
    }
  }
  __syncthreads();

  int wave = t >> 5, lane = t & 31;
  int ln = lane & 15, kh2 = lane >> 4;
  int rowbase = tile * 128 + wave * 16;

  // per-lane collision bitmask: column s = n*16+ln collides if its sample's
  // sorted position lives in this workgroup's 256-block
  int cb = 0;
#pragma unroll
  for (int n = 0; n < 16; n++) {
    int si = smp[bh * SMP + n * 16 + ln];
    cb |= ((si >> 8) == blk) ? (1 << n) : 0;
  }

  const _Float16* qrow = qs + (bhN + rowbase + ln) * DD;
  v16h qa0 = load_frag(qrow + kh2 * 8);
  v16h qa1 = load_frag(qrow + 32 + kh2 * 8);

  v8f S[16];
#pragma unroll
  for (int n = 0; n < 16; n++) {
    v16h b0 = load_frag(&Kt[(n * 16 + ln) * DD + kh2 * 8]);
    v16h b1 = load_frag(&Kt[(n * 16 + ln) * DD + 32 + kh2 * 8]);
    v8f acc = {};
    acc = WMMA_F16(qa0, b0, acc);
    acc = WMMA_F16(qa1, b1, acc);
    S[n] = acc;
  }

  // scale + collision bias (float-min style), then softmax
#pragma unroll
  for (int n = 0; n < 16; n++) {
    float b = ((cb >> n) & 1) ? -3.0e38f : 0.0f;
#pragma unroll
    for (int r = 0; r < 8; r++) S[n][r] = S[n][r] * SCALE_ + b;
  }
  float mx[8], sm[8], lse2[8];
#pragma unroll
  for (int r = 0; r < 8; r++) {
    float m = -3.0e38f;
#pragma unroll
    for (int n = 0; n < 16; n++) m = fmaxf(m, S[n][r]);
#pragma unroll
    for (int o = 1; o < 16; o <<= 1) m = fmaxf(m, __shfl_xor(m, o, 32));
    mx[r] = m;
  }
#pragma unroll
  for (int r = 0; r < 8; r++) {
    float s = 0.0f;
#pragma unroll
    for (int n = 0; n < 16; n++) {
      float p = __expf(S[n][r] - mx[r]);
      S[n][r] = p;
      s += p;
    }
#pragma unroll
    for (int o = 1; o < 16; o <<= 1) s += __shfl_xor(s, o, 32);
    sm[r] = s;
    lse2[r] = mx[r] + __logf(s) + LOG32_;  // + log(N/SAMPLE)
  }

  __syncthreads();
  _Float16* Pst = Kt + wave * 512;

  v8f O[4] = {};
#pragma unroll
  for (int kc = 0; kc < 8; kc++) {
#pragma unroll
    for (int u = 0; u < 2; u++) {
      int n = kc * 2 + u;
#pragma unroll
      for (int r = 0; r < 8; r++)
        Pst[(r + kh2 * 8) * 32 + u * 16 + ln] = (_Float16)S[n][r];
    }
    __builtin_amdgcn_wave_barrier();
    v16h pa = load_frag(&Pst[ln * 32 + kh2 * 8]);
#pragma unroll
    for (int nv = 0; nv < 4; nv++) {
      v16h vb = load_frag(&Vt[(nv * 16 + ln) * SMP + kc * 32 + kh2 * 8]);
      O[nv] = WMMA_F16(pa, vb, O[nv]);
    }
    __builtin_amdgcn_wave_barrier();
  }

  // merge with block attention (already in d_out at un-sorted rows)
  float w1[8], w2[8];
  int ur[8];
#pragma unroll
  for (int r = 0; r < 8; r++) {
    long long srow = bhN + rowbase + kh2 * 8 + r;
    float l1 = lseW[srow];
    float c = 1.0f / (1.0f + __expf(lse2[r] - l1));  // sigmoid(l1 - l2)
    w1[r] = c;
    w2[r] = (1.0f - c) / sm[r];
    ur[r] = qi[srow];
  }
#pragma unroll
  for (int nv = 0; nv < 4; nv++)
#pragma unroll
    for (int r = 0; r < 8; r++) {
      long long a = (bhN + ur[r]) * DD + nv * 16 + ln;
      out[a] = w1[r] * out[a] + w2[r] * O[nv][r];
    }
}

// ---------------------------------------------------------------------------
extern "C" void kernel_launch(void* const* d_in, const int* in_sizes, int n_in,
                              void* d_out, int out_size, void* d_ws, size_t ws_size,
                              hipStream_t stream) {
  const float* query = (const float*)d_in[0];
  const float* key   = (const float*)d_in[1];
  const float* value = (const float*)d_in[2];
  const float* pd    = (const float*)d_in[3];
  const int*   smp   = (const int*)d_in[4];
  float* out = (float*)d_out;

  char* ws = (char*)d_ws;
  int*   qh   = (int*)(ws + 0);
  int*   kh   = (int*)(ws + (size_t)BHN_ * 4);
  int*   qi   = (int*)(ws + (size_t)BHN_ * 8);
  int*   ki   = (int*)(ws + (size_t)BHN_ * 12);
  float* lseW = (float*)(ws + (size_t)BHN_ * 16);
  _Float16* qs = (_Float16*)(ws + (size_t)BHN_ * 20);
  _Float16* ks = (_Float16*)(ws + (size_t)BHN_ * 20 + (size_t)BHN_ * DD * 2);
  _Float16* vs = (_Float16*)(ws + (size_t)BHN_ * 20 + (size_t)BHN_ * DD * 4);

  hash_kernel<<<(2 * BHN_) / 256, 256, 0, stream>>>(query, key, pd, qh, kh);
  csort_kernel<<<2 * BH_, 128, 0, stream>>>(qh, kh, qi, ki);
  gather_kernel<<<(3 * BHN_) / 256, 256, 0, stream>>>(query, key, value, qi, ki, qs, ks, vs);
  blockattn_kernel<<<BH_ * (NN / BLK) * 2, 256, 0, stream>>>(qs, ks, vs, qi, lseW, out);
  residual_kernel<<<BH_ * (NN / 128), 256, 0, stream>>>(qs, ks, vs, qi, smp, lseW, out);
}
